// SelectMolAttachment_88553635709674
// MI455X (gfx1250) — matrix operand order repaired and checked by reading, not compiled, not run on hardware
//
#include <hip/hip_runtime.h>
#include <hip/hip_bf16.h>

typedef unsigned short u16;
typedef unsigned int   u32;
typedef __attribute__((ext_vector_type(16))) __bf16 v16bf;
typedef __attribute__((ext_vector_type(8)))  float  v8f;

union FragB { v16bf v; u32 u[8]; };

__device__ __forceinline__ u16 f2bf(float f) {
  u32 u = __float_as_uint(f);
  u += 0x7fffu + ((u >> 16) & 1u);          // round-to-nearest-even
  return (u16)(u >> 16);
}
__device__ __forceinline__ float bf2f(u16 h) {
  return __uint_as_float(((u32)h) << 16);
}
__device__ __forceinline__ u32 pack2(float a, float b) {
  return (u32)f2bf(a) | ((u32)f2bf(b) << 16);
}

struct Tile { uint4 x0, x1; };            // one thread's 16-u16 x2 staging chunk

__device__ __forceinline__ void store_tile(u16* dst, const Tile& t) {
  ((uint4*)dst)[0] = t.x0;
  ((uint4*)dst)[1] = t.x1;
}

// ---------------------------------------------------------------------------
// Generic bf16 WMMA GEMM:  C[M x N] = act(A[M x K] @ Wt^T + bias)
//   Wt stored transposed+padded: [Npad rows][K] bf16 (Npad = gridDim.x*128)
//   AMODE 0: A is bf16 row-major (row stride K)
//   AMODE 1: A row r -> molecule b=r>>6, slot m=r&63; valid slots gathered
//            from f32 gsrc (row stride K) via offs/cnts, pad slots = 0
//   ACT: 0 none, 1 relu, 2 sigmoid ; OUTF: 0 bf16 out, 1 f32 out
// Tiling: 128x128 block, BK=32, 256 threads (8 waves), wave = 32x64 subtile.
// Double-buffered LDS: global loads for step k+1 issue before the WMMA block
// of step k (latency overlap); one workgroup barrier per K-step.
// ---------------------------------------------------------------------------
#define LDT 40   // LDS tile row stride in u16 (80B: 16B-aligned, conflict-light)

template<int AMODE, int ACT, int OUTF>
__global__ __launch_bounds__(256)
void gemm_bf16_wmma(const void* __restrict__ Aq,
                    const u16*  __restrict__ Wt,
                    const float* __restrict__ biasp,   // padded to Npad
                    void* __restrict__ Cq,
                    int M, int N, int K, int ldc,
                    const int* __restrict__ offs,
                    const int* __restrict__ cnts,
                    const float* __restrict__ gsrc)
{
  __shared__ __align__(16) u16 sA[2][128 * LDT];
  __shared__ __align__(16) u16 sB[2][128 * LDT];

  const int tid  = threadIdx.x;
  const int lane = tid & 31;
  const int w    = tid >> 5;
  const int wr   = w >> 1;          // 0..3 -> 32-row group
  const int wc   = w & 1;           // 0..1 -> 64-col group
  const int bm0  = blockIdx.y * 128;
  const int bn0  = blockIdx.x * 128;

  const int tr  = tid >> 1;         // staging: tile row 0..127
  const int seg = tid & 1;          // staging: 16-elem segment

  // gather precompute (AMODE 1)
  int gvalid = 0; const float* grow = nullptr;
  if (AMODE == 1) {
    const int r = bm0 + tr;
    const int mol = r >> 6, m = r & 63;
    gvalid = (m < cnts[mol]);
    if (gvalid) grow = gsrc + (size_t)(offs[mol] + m) * K;
  }
  const u16* arow = (AMODE == 0) ? ((const u16*)Aq + (size_t)(bm0 + tr) * K) : nullptr;
  const u16* brow = Wt + (size_t)(bn0 + tr) * K;

  auto loadA = [&](int k0) -> Tile {
    Tile t;
    if (AMODE == 0) {
      const u16* p = arow + k0 + seg * 16;
      t.x0 = ((const uint4*)p)[0];
      t.x1 = ((const uint4*)p)[1];
    } else {
      t.x0 = uint4{0,0,0,0}; t.x1 = uint4{0,0,0,0};
      if (gvalid) {
        const float4* p4 = (const float4*)(grow + k0 + seg * 16);
        float4 f0 = p4[0], f1 = p4[1], f2 = p4[2], f3 = p4[3];
        t.x0.x = pack2(f0.x, f0.y); t.x0.y = pack2(f0.z, f0.w);
        t.x0.z = pack2(f1.x, f1.y); t.x0.w = pack2(f1.z, f1.w);
        t.x1.x = pack2(f2.x, f2.y); t.x1.y = pack2(f2.z, f2.w);
        t.x1.z = pack2(f3.x, f3.y); t.x1.w = pack2(f3.z, f3.w);
      }
    }
    return t;
  };
  auto loadB = [&](int k0) -> Tile {
    Tile t;
    const u16* p = brow + k0 + seg * 16;
    t.x0 = ((const uint4*)p)[0];
    t.x1 = ((const uint4*)p)[1];
    return t;
  };

  const v8f vz = {0.f,0.f,0.f,0.f,0.f,0.f,0.f,0.f};
  v8f acc[2][4];
  #pragma unroll
  for (int i = 0; i < 2; i++)
    #pragma unroll
    for (int j = 0; j < 4; j++) acc[i][j] = vz;

  const int lr  = lane & 15;
  const int kh  = (lane >> 4) << 3;   // A frag: 0 or 8
  const int kh2 = (lane >> 4) << 4;   // unused: 0 or 16

  // ---- prologue: stage tile 0 ----
  {
    Tile ta = loadA(0), tb = loadB(0);
    store_tile(&sA[0][tr * LDT + seg * 16], ta);
    store_tile(&sB[0][tr * LDT + seg * 16], tb);
  }
  __syncthreads();

  int cur = 0;
  for (int k0 = 0; k0 < K; k0 += 32) {
    const bool has_next = (k0 + 32) < K;
    Tile na, nb;
    if (has_next) {
      na = loadA(k0 + 32);                         // overlap with WMMA below
      nb = loadB(k0 + 32);
      if (k0 + 64 < K) {                           // warm WGP$/L2 two steps out
        if (AMODE == 0)      __builtin_prefetch(arow + k0 + 64 + seg * 16, 0, 0);
        else if (gvalid)     __builtin_prefetch(grow + k0 + 64 + seg * 16, 0, 0);
        __builtin_prefetch(brow + k0 + 64 + seg * 16, 0, 0);
      }
    }

    // ---- compute from buffers[cur] ----
    const u16* As = sA[cur];
    const u16* Bs = sB[cur];
    FragB af[2];
    #pragma unroll
    for (int i = 0; i < 2; i++) {
      const u16* p = As + (wr * 32 + i * 16 + lr) * LDT;
      #pragma unroll
      for (int j = 0; j < 4; j++) {
        af[i].u[j]     = *(const u32*)(p + kh + 2 * j);
        af[i].u[4 + j] = *(const u32*)(p + 16 + kh + 2 * j);
      }
    }
    #pragma unroll
    for (int j = 0; j < 4; j++) {
      FragB bfg;
      const u16* p = Bs + (wc * 64 + j * 16 + lr) * LDT;
      #pragma unroll
      for (int t = 0; t < 4; t++) { bfg.u[t] = *(const u32*)(p + kh + 2 * t); bfg.u[4 + t] = *(const u32*)(p + 16 + kh + 2 * t); }
      acc[0][j] = __builtin_amdgcn_wmma_f32_16x16x32_bf16(false, af[0].v, false, bfg.v,
                                                          (short)0, acc[0][j], false, false);
      acc[1][j] = __builtin_amdgcn_wmma_f32_16x16x32_bf16(false, af[1].v, false, bfg.v,
                                                          (short)0, acc[1][j], false, false);
    }

    if (has_next) {
      const int nxt = cur ^ 1;
      store_tile(&sA[nxt][tr * LDT + seg * 16], na);
      store_tile(&sB[nxt][tr * LDT + seg * 16], nb);
    }
    __syncthreads();
    cur ^= 1;
  }

  // ---- epilogue: bias + activation + store ----
  #pragma unroll
  for (int i = 0; i < 2; i++) {
    const int mbase = bm0 + wr * 32 + i * 16 + ((lane >> 4) << 3);
    #pragma unroll
    for (int j = 0; j < 4; j++) {
      const int n = bn0 + wc * 64 + j * 16 + lr;
      if (n < N) {
        const float bv = biasp[n];
        #pragma unroll
        for (int r = 0; r < 8; r++) {
          float vv = acc[i][j][r] + bv;
          if (ACT == 1) vv = fmaxf(vv, 0.f);
          if (ACT == 2) vv = 1.f / (1.f + __expf(-vv));
          const size_t o = (size_t)(mbase + r) * ldc + n;
          if (OUTF == 0) ((u16*)Cq)[o] = f2bf(vv);
          else           ((float*)Cq)[o] = vv;
        }
      }
    }
  }
}

// ---------------------------------------------------------------------------
// Per-molecule attention: S = K Q^T (64x64), masked softmax, corr = A V
// K,Q,V: bf16 row-major [64][128] per molecule. 1 workgroup (256t) / molecule.
// ---------------------------------------------------------------------------
__global__ __launch_bounds__(256)
void attn_kernel(const u16* __restrict__ Kmat, const u16* __restrict__ Qmat,
                 const u16* __restrict__ Vmat, u16* __restrict__ corr,
                 const int* __restrict__ cnts)
{
  __shared__ __align__(16) u16   Ksh[64 * 136];
  __shared__ __align__(16) u16   Vt [128 * 72];   // V transposed: [feat][atom]
  __shared__ __align__(16) float Ssh[64 * 65];
  __shared__ __align__(16) u16   Ash[64 * 72];    // attn weights bf16

  const int b    = blockIdx.x;
  const int tid  = threadIdx.x;
  const int lane = tid & 31;
  const int w    = tid >> 5;
  const int wr   = w >> 1, wc = w & 1;
  const int lr   = lane & 15;
  const int kh   = (lane >> 4) << 3;
  const int kh2  = (lane >> 4) << 4;
  const size_t base = (size_t)b * 64 * 128;

  for (int c = tid; c < 64 * 8; c += 256) {       // K rows, 16-u16 chunks
    int row = c >> 3, sg = c & 7;
    ((uint4*)(Ksh + row * 136 + sg * 16))[0] =
        ((const uint4*)(Kmat + base + row * 128 + sg * 16))[0];
  }
  for (int e = tid; e < 64 * 128; e += 256) {     // V transposed into LDS
    int m = e >> 7, n = e & 127;
    Vt[n * 72 + m] = Vmat[base + e];
  }
  __syncthreads();

  // Phase 1: scores S = K @ Q^T  (Q B-fragments straight from L2)
  {
    const v8f vz = {0.f,0.f,0.f,0.f,0.f,0.f,0.f,0.f};
    v8f acc[2] = { vz, vz };
    for (int k0 = 0; k0 < 128; k0 += 32) {
      FragB a;
      const u16* pa = Ksh + (wr * 16 + lr) * 136 + k0;
      #pragma unroll
      for (int j = 0; j < 4; j++) {
        a.u[j]     = *(const u32*)(pa + kh + 2 * j);
        a.u[4 + j] = *(const u32*)(pa + 16 + kh + 2 * j);
      }
      #pragma unroll
      for (int j = 0; j < 2; j++) {
        FragB bq;
        const u16* pb = Qmat + base + (size_t)(wc * 32 + j * 16 + lr) * 128 + k0;
        #pragma unroll
        for (int t = 0; t < 4; t++) { bq.u[t] = *(const u32*)(pb + kh + 2 * t); bq.u[4 + t] = *(const u32*)(pb + 16 + kh + 2 * t); }
        acc[j] = __builtin_amdgcn_wmma_f32_16x16x32_bf16(false, a.v, false, bq.v,
                                                         (short)0, acc[j], false, false);
      }
    }
    #pragma unroll
    for (int j = 0; j < 2; j++) {
      int col   = wc * 32 + j * 16 + lr;
      int rbase = wr * 16 + ((lane >> 4) << 3);
      #pragma unroll
      for (int r = 0; r < 8; r++) Ssh[(rbase + r) * 65 + col] = acc[j][r];
    }
  }
  __syncthreads();

  // Phase 2: masked softmax per row (matches reference -1e9 masking semantics)
  if (tid < 64) {
    const int m = tid;
    const int cnt = cnts[b];
    const float sc = 0.08838834764831845f;       // 1/sqrt(128)
    if (m < cnt) {
      float mx = -3.4e38f;
      for (int n = 0; n < cnt; n++) mx = fmaxf(mx, Ssh[m * 65 + n] * sc);
      float s = 0.f;
      for (int n = 0; n < cnt; n++) {
        float e = __expf(Ssh[m * 65 + n] * sc - mx);
        Ssh[m * 65 + n] = e; s += e;
      }
      float inv = 1.f / s;
      for (int n = 0; n < 64; n++)
        Ash[m * 72 + n] = (n < cnt) ? f2bf(Ssh[m * 65 + n] * inv) : (u16)0;
    } else if (m < 50) {                         // fully-masked slot: uniform 1/50
      for (int n = 0; n < 64; n++) Ash[m * 72 + n] = (n < 50) ? f2bf(0.02f) : (u16)0;
    } else {
      for (int n = 0; n < 64; n++) Ash[m * 72 + n] = 0;
    }
  }
  __syncthreads();

  // Phase 3: corr = A @ V  (64 x 128)
  {
    const v8f vz = {0.f,0.f,0.f,0.f,0.f,0.f,0.f,0.f};
    v8f acc[4] = { vz, vz, vz, vz };
    for (int k0 = 0; k0 < 64; k0 += 32) {
      FragB a;
      const u16* pa = Ash + (wr * 16 + lr) * 72 + k0;
      #pragma unroll
      for (int j = 0; j < 4; j++) {
        a.u[j]     = *(const u32*)(pa + kh + 2 * j);
        a.u[4 + j] = *(const u32*)(pa + 16 + kh + 2 * j);
      }
      #pragma unroll
      for (int j = 0; j < 4; j++) {
        FragB bv;
        const u16* pb = Vt + (wc * 64 + j * 16 + lr) * 72 + k0;
        #pragma unroll
        for (int t = 0; t < 4; t++) { bv.u[t] = *(const u32*)(pb + kh + 2 * t); bv.u[4 + t] = *(const u32*)(pb + 16 + kh + 2 * t); }
        acc[j] = __builtin_amdgcn_wmma_f32_16x16x32_bf16(false, a.v, false, bv.v,
                                                         (short)0, acc[j], false, false);
      }
    }
    #pragma unroll
    for (int j = 0; j < 4; j++) {
      int col   = wc * 64 + j * 16 + lr;
      int rbase = wr * 16 + ((lane >> 4) << 3);
      #pragma unroll
      for (int r = 0; r < 8; r++)
        corr[base + (size_t)(rbase + r) * 128 + col] = f2bf(acc[j][r]);
    }
  }
}

// ---------------------------------------------------------------------------
// Glue kernels
// ---------------------------------------------------------------------------
__global__ void offsets_kernel(const int* __restrict__ cnts, int* __restrict__ offs, int B) {
  if (blockIdx.x == 0 && threadIdx.x == 0) {
    int a = 0;
    for (int i = 0; i < B; i++) { offs[i] = a; a += cnts[i]; }
  }
}

// W is f32 row-major [Kreal][Nreal]; Wt is bf16 [Npad][Kpad], zero-padded
__global__ void convert_weight(const float* __restrict__ W, u16* __restrict__ Wt,
                               int Kreal, int Nreal, int Kpad, int Npad) {
  int idx = blockIdx.x * blockDim.x + threadIdx.x;
  if (idx >= Kpad * Npad) return;
  int n = idx / Kpad, k = idx - n * Kpad;
  float v = (k < Kreal && n < Nreal) ? W[(size_t)k * Nreal + n] : 0.f;
  Wt[(size_t)n * Kpad + k] = f2bf(v);
}

__global__ void pad_bias(const float* __restrict__ bsrc, float* __restrict__ bdst,
                         int nreal, int npad) {
  int i = blockIdx.x * blockDim.x + threadIdx.x;
  if (i < npad) bdst[i] = (i < nreal) ? bsrc[i] : 0.f;
}

__global__ void conv_f32_bf16(const float* __restrict__ src, u16* __restrict__ dst, int n) {
  int i = blockIdx.x * blockDim.x + threadIdx.x;
  if (i < n) dst[i] = f2bf(src[i]);
}

__global__ __launch_bounds__(64)
void logits_kernel(const u16* __restrict__ K2, const u16* __restrict__ Q2,
                   u16* __restrict__ Lbf) {
  int b = blockIdx.x, m = threadIdx.x;
  const u16* kr = K2 + ((size_t)b * 64 + m) * 128;
  const u16* qr = Q2 + (size_t)b * 128;
  float s = 0.f;
  for (int k = 0; k < 128; k++) s += bf2f(kr[k]) * bf2f(qr[k]);
  s *= 0.08838834764831845f;
  Lbf[(size_t)b * 64 + m] = (m < 50) ? f2bf(s) : (u16)0;
}

__global__ __launch_bounds__(64)
void scatter_kernel(const float* __restrict__ sel, const int* __restrict__ offs,
                    const int* __restrict__ cnts, float* __restrict__ out) {
  int b = blockIdx.x, m = threadIdx.x;
  if (m < cnts[b]) out[offs[b] + m] = sel[(size_t)b * 64 + m];
}

// ---------------------------------------------------------------------------
extern "C" void kernel_launch(void* const* d_in, const int* in_sizes, int n_in,
                              void* d_out, int out_size, void* d_ws, size_t ws_size,
                              hipStream_t stream)
{
  (void)in_sizes; (void)n_in; (void)out_size; (void)ws_size;
  const float* mol_a  = (const float*)d_in[0];
  const float* node_h = (const float*)d_in[1];
  const int*   cnts   = (const int*)d_in[2];
  const float* wq1 = (const float*)d_in[3];  const float* bq1  = (const float*)d_in[4];
  const float* wq2 = (const float*)d_in[5];  const float* bq2  = (const float*)d_in[6];
  const float* wk1 = (const float*)d_in[7];  const float* bk1  = (const float*)d_in[8];
  const float* wk2 = (const float*)d_in[9];  const float* bk2  = (const float*)d_in[10];
  const float* wv1 = (const float*)d_in[11]; const float* bv1  = (const float*)d_in[12];
  const float* wv2 = (const float*)d_in[13]; const float* bv2  = (const float*)d_in[14];
  const float* uk1 = (const float*)d_in[15]; const float* ubk1 = (const float*)d_in[16];
  const float* uk2 = (const float*)d_in[17]; const float* ubk2 = (const float*)d_in[18];
  const float* uq1 = (const float*)d_in[19]; const float* ubq1 = (const float*)d_in[20];
  const float* uq2 = (const float*)d_in[21]; const float* ubq2 = (const float*)d_in[22];
  const float* cs1 = (const float*)d_in[23]; const float* bcs1 = (const float*)d_in[24];
  const float* cs2 = (const float*)d_in[25]; const float* bcs2 = (const float*)d_in[26];

  const int Bm = 2048;
  const int R  = Bm * 64;          // 131072 padded rows

  char* ws = (char*)d_ws;
  size_t off = 0;
  auto alloc = [&](size_t bytes) -> char* {
    char* p = ws + off;
    off = (off + bytes + 255) & ~(size_t)255;
    return p;
  };
  int* offs   = (int*)alloc(Bm * 4);
  u16* wq1t = (u16*)alloc((size_t)256 * 512 * 2);
  u16* wk1t = (u16*)alloc((size_t)256 * 512 * 2);
  u16* wv1t = (u16*)alloc((size_t)256 * 512 * 2);
  u16* wq2t = (u16*)alloc((size_t)128 * 256 * 2);
  u16* wk2t = (u16*)alloc((size_t)128 * 256 * 2);
  u16* wv2t = (u16*)alloc((size_t)128 * 256 * 2);
  u16* uk1t = (u16*)alloc((size_t)256 * 128 * 2);
  u16* uk2t = (u16*)alloc((size_t)128 * 256 * 2);
  u16* uq1t = (u16*)alloc((size_t)256 * 512 * 2);
  u16* uq2t = (u16*)alloc((size_t)128 * 256 * 2);
  u16* cs1t = (u16*)alloc((size_t)128 * 64 * 2);
  u16* cs2t = (u16*)alloc((size_t)128 * 128 * 2);
  float* bq1p  = (float*)alloc(256 * 4); float* bq2p  = (float*)alloc(128 * 4);
  float* bk1p  = (float*)alloc(256 * 4); float* bk2p  = (float*)alloc(128 * 4);
  float* bv1p  = (float*)alloc(256 * 4); float* bv2p  = (float*)alloc(128 * 4);
  float* ubk1p = (float*)alloc(256 * 4); float* ubk2p = (float*)alloc(128 * 4);
  float* ubq1p = (float*)alloc(256 * 4); float* ubq2p = (float*)alloc(128 * 4);
  float* bcs1p = (float*)alloc(128 * 4); float* bcs2p = (float*)alloc(128 * 4);
  u16*  molAb = (u16*)alloc((size_t)2048 * 512 * 2);
  u16*  Q2b   = (u16*)alloc((size_t)2048 * 128 * 2);
  u16*  Lbf   = (u16*)alloc((size_t)2048 * 64 * 2);
  float* selb = (float*)alloc((size_t)2048 * 64 * 4);
  u16*  Kmat  = (u16*)alloc((size_t)R * 128 * 2);
  u16*  Qmat  = (u16*)alloc((size_t)R * 128 * 2);
  u16*  Vmat  = (u16*)alloc((size_t)R * 128 * 2);
  u16*  corr  = (u16*)alloc((size_t)R * 128 * 2);
  u16*  Hbuf  = (u16*)alloc((size_t)R * 256 * 2);
  u16*  K2    = Kmat;   // Kmat is dead after attention; reuse for K2
  u16*  Gm    = Hbuf;   // Hbuf is dead after G4; reuse for sel hidden

  // --- prep: offsets, weight/bias conversion ---
  offsets_kernel<<<1, 1, 0, stream>>>(cnts, offs, Bm);
  convert_weight<<<(512*256+255)/256, 256, 0, stream>>>(wq1, wq1t, 512, 256, 512, 256);
  convert_weight<<<(512*256+255)/256, 256, 0, stream>>>(wk1, wk1t, 512, 256, 512, 256);
  convert_weight<<<(512*256+255)/256, 256, 0, stream>>>(wv1, wv1t, 512, 256, 512, 256);
  convert_weight<<<(256*128+255)/256, 256, 0, stream>>>(wq2, wq2t, 256, 128, 256, 128);
  convert_weight<<<(256*128+255)/256, 256, 0, stream>>>(wk2, wk2t, 256, 128, 256, 128);
  convert_weight<<<(256*128+255)/256, 256, 0, stream>>>(wv2, wv2t, 256, 128, 256, 128);
  convert_weight<<<(128*256+255)/256, 256, 0, stream>>>(uk1, uk1t, 128, 256, 128, 256);
  convert_weight<<<(256*128+255)/256, 256, 0, stream>>>(uk2, uk2t, 256, 128, 256, 128);
  convert_weight<<<(512*256+255)/256, 256, 0, stream>>>(uq1, uq1t, 512, 256, 512, 256);
  convert_weight<<<(256*128+255)/256, 256, 0, stream>>>(uq2, uq2t, 256, 128, 256, 128);
  convert_weight<<<(64*128+255)/256,  256, 0, stream>>>(cs1, cs1t, 50, 128, 64, 128);
  convert_weight<<<(128*128+255)/256, 256, 0, stream>>>(cs2, cs2t, 128, 50, 128, 128);
  pad_bias<<<1, 256, 0, stream>>>(bq1,  bq1p,  256, 256);
  pad_bias<<<1, 256, 0, stream>>>(bq2,  bq2p,  128, 128);
  pad_bias<<<1, 256, 0, stream>>>(bk1,  bk1p,  256, 256);
  pad_bias<<<1, 256, 0, stream>>>(bk2,  bk2p,  128, 128);
  pad_bias<<<1, 256, 0, stream>>>(bv1,  bv1p,  256, 256);
  pad_bias<<<1, 256, 0, stream>>>(bv2,  bv2p,  128, 128);
  pad_bias<<<1, 256, 0, stream>>>(ubk1, ubk1p, 256, 256);
  pad_bias<<<1, 256, 0, stream>>>(ubk2, ubk2p, 128, 128);
  pad_bias<<<1, 256, 0, stream>>>(ubq1, ubq1p, 256, 256);
  pad_bias<<<1, 256, 0, stream>>>(ubq2, ubq2p, 128, 128);
  pad_bias<<<1, 256, 0, stream>>>(bcs1, bcs1p, 128, 128);
  pad_bias<<<1, 256, 0, stream>>>(bcs2, bcs2p, 50, 128);
  conv_f32_bf16<<<(2048*512+255)/256, 256, 0, stream>>>(mol_a, molAb, 2048 * 512);

  // --- K/Q/V MLPs (ragged gather fused into layer 1) ---
  gemm_bf16_wmma<1,1,0><<<dim3(2, R/128), 256, 0, stream>>>(nullptr, wk1t, bk1p, Hbuf, R, 256, 512, 256, offs, cnts, node_h);
  gemm_bf16_wmma<0,0,0><<<dim3(1, R/128), 256, 0, stream>>>(Hbuf, wk2t, bk2p, Kmat, R, 128, 256, 128, nullptr, nullptr, nullptr);
  gemm_bf16_wmma<1,1,0><<<dim3(2, R/128), 256, 0, stream>>>(nullptr, wq1t, bq1p, Hbuf, R, 256, 512, 256, offs, cnts, node_h);
  gemm_bf16_wmma<0,0,0><<<dim3(1, R/128), 256, 0, stream>>>(Hbuf, wq2t, bq2p, Qmat, R, 128, 256, 128, nullptr, nullptr, nullptr);
  gemm_bf16_wmma<1,1,0><<<dim3(2, R/128), 256, 0, stream>>>(nullptr, wv1t, bv1p, Hbuf, R, 256, 512, 256, offs, cnts, node_h);
  gemm_bf16_wmma<0,0,0><<<dim3(1, R/128), 256, 0, stream>>>(Hbuf, wv2t, bv2p, Vmat, R, 128, 256, 128, nullptr, nullptr, nullptr);

  // --- Q2 = mlp2(mol_a_reprs) ---
  gemm_bf16_wmma<0,1,0><<<dim3(2, 16), 256, 0, stream>>>(molAb, uq1t, ubq1p, Hbuf, 2048, 256, 512, 256, nullptr, nullptr, nullptr);
  gemm_bf16_wmma<0,0,0><<<dim3(1, 16), 256, 0, stream>>>(Hbuf, uq2t, ubq2p, Q2b, 2048, 128, 256, 128, nullptr, nullptr, nullptr);

  // --- attention per molecule ---
  attn_kernel<<<2048, 256, 0, stream>>>(Kmat, Qmat, Vmat, corr, cnts);

  // --- K2 = mlp2(corr) ---
  gemm_bf16_wmma<0,1,0><<<dim3(2, R/128), 256, 0, stream>>>(corr, uk1t, ubk1p, Hbuf, R, 256, 128, 256, nullptr, nullptr, nullptr);
  gemm_bf16_wmma<0,0,0><<<dim3(1, R/128), 256, 0, stream>>>(Hbuf, uk2t, ubk2p, K2, R, 128, 256, 128, nullptr, nullptr, nullptr);

  // --- logits, selection MLP, scatter ---
  logits_kernel<<<2048, 64, 0, stream>>>(K2, Q2b, Lbf);
  gemm_bf16_wmma<0,1,0><<<dim3(1, 16), 256, 0, stream>>>(Lbf, cs1t, bcs1p, Gm, 2048, 128, 64, 128, nullptr, nullptr, nullptr);
  gemm_bf16_wmma<0,2,1><<<dim3(1, 16), 256, 0, stream>>>(Gm, cs2t, bcs2p, selb, 2048, 64, 128, 64, nullptr, nullptr, nullptr);
  scatter_kernel<<<2048, 64, 0, stream>>>(selb, offs, cnts, (float*)d_out);
}
